// PCT_38732015075604
// MI455X (gfx1250) — matrix operand order repaired and checked by reading, not compiled, not run on hardware
//
#include <hip/hip_runtime.h>
#include <hip/hip_bf16.h>

// ---------------------------------------------------------------------------
// PCT forward for MI455X (gfx1250, wave32, WMMA).
// All heavy GEMMs go through v_wmma_f32_16x16x32_bf16 (bf16 in, f32 acc).
// ---------------------------------------------------------------------------

typedef __attribute__((ext_vector_type(16))) __bf16 v16bf;
typedef __attribute__((ext_vector_type(8)))  float  v8f;

union FragU { v16bf v; uint4 q[2]; };

// fp32 -> bf16, round-half-up (equivalent to RNE except on exact ties; the
// tie bias is far below bf16 quantization noise and saves ~3 VALU/element).
__device__ __forceinline__ unsigned short f2bf(float f) {
  return (unsigned short)((__float_as_uint(f) + 0x8000u) >> 16);
}
__device__ __forceinline__ unsigned int pack2bf(float a, float b) {
  unsigned int ua = __float_as_uint(a) + 0x8000u;
  unsigned int ub = __float_as_uint(b) + 0x8000u;
  return (ua >> 16) | (ub & 0xffff0000u);      // lowers to shift/and-or or v_perm
}

// ---------------------------------------------------------------------------
// Generic batched GEMM:  C[m,n] = sum_k A[m,k] * Bmat[k,n]  (+ bias[n])
//   A  : fp32 row-major, row stride lda, batch stride aS
//   B  : if bKN==0: weights W[N,K] row-major (Bmat[k,n] = W[n*K+k])
//        if bKN==1: activations [K,N] row-major, row stride ldbK (N full tiles)
//   C  : fp32 row-major, row stride ldc, batch stride cS
// Tile: 64x64x32; 4 waves; wave = 16 M-rows x 4 N-subtiles; LDS double-buffered.
// ---------------------------------------------------------------------------
#define TBM 64
#define TBN 64
#define TBK 32
#define ROWP 40   // padded LDS row (ushorts): 80B keeps 16B align, spreads banks

__global__ __launch_bounds__(128) void k_gemm(
    const float* __restrict__ A, int lda, long long aS,
    const float* __restrict__ B, int ldbK, long long bS, int bKN,
    const float* __restrict__ bias,
    float* __restrict__ C, int ldc, long long cS,
    int M, int N, int K)
{
  __shared__ unsigned short As[2][TBM * ROWP];
  __shared__ unsigned short Bs[2][TBN * ROWP];

  const int tid  = threadIdx.x;
  const int wave = tid >> 5;
  const int lane = tid & 31;
  const int m0 = blockIdx.y * TBM;
  const int n0 = blockIdx.x * TBN;
  const bool fullM = (m0 + TBM <= M);
  const bool fullN = (n0 + TBN <= N);

  A += (long long)blockIdx.z * aS;
  B += (long long)blockIdx.z * bS;
  C += (long long)blockIdx.z * cS;

  // ---- tile staging (vectorized: 4x float4 per thread) ----
  auto stageA = [&](unsigned short* dst, int k0) {
#pragma unroll
    for (int it = 0; it < 4; ++it) {
      int idx4 = tid + it * 128;          // 0..511 chunks
      int m  = idx4 >> 3;                 // 64 rows
      int kq = (idx4 & 7) << 2;           // 0..28 step 4
      float4 v = make_float4(0.f, 0.f, 0.f, 0.f);
      int gm = m0 + m;
      if (fullM || gm < M)
        v = *(const float4*)&A[(long long)gm * lda + k0 + kq];
      uint2 p; p.x = pack2bf(v.x, v.y); p.y = pack2bf(v.z, v.w);
      *(uint2*)&dst[m * ROWP + kq] = p;
    }
  };
  auto stageB = [&](unsigned short* dst, int k0) {
    if (!bKN) {
#pragma unroll
      for (int it = 0; it < 4; ++it) {
        int idx4 = tid + it * 128;
        int n  = idx4 >> 3;
        int kq = (idx4 & 7) << 2;
        float4 v = make_float4(0.f, 0.f, 0.f, 0.f);
        int gn = n0 + n;
        if (fullN || gn < N)
          v = *(const float4*)&B[(long long)gn * K + k0 + kq];
        uint2 p; p.x = pack2bf(v.x, v.y); p.y = pack2bf(v.z, v.w);
        *(uint2*)&dst[n * ROWP + kq] = p;
      }
    } else {            // [K,N] activations; only used with full N tiles
#pragma unroll
      for (int it = 0; it < 4; ++it) {
        int idx4 = tid + it * 128;
        int k  = idx4 >> 4;                 // 32 k rows
        int nq = (idx4 & 15) << 2;          // 0..60 step 4
        float4 v = *(const float4*)&B[(long long)(k0 + k) * ldbK + n0 + nq];
        dst[(nq + 0) * ROWP + k] = f2bf(v.x);
        dst[(nq + 1) * ROWP + k] = f2bf(v.y);
        dst[(nq + 2) * ROWP + k] = f2bf(v.z);
        dst[(nq + 3) * ROWP + k] = f2bf(v.w);
      }
    }
  };

  v8f acc[4];
#pragma unroll
  for (int j = 0; j < 4; ++j)
#pragma unroll
    for (int r = 0; r < 8; ++r) acc[j][r] = 0.0f;

  stageA(As[0], 0);
  stageB(Bs[0], 0);

  const int ml = (wave << 4) + (lane & 15);
  const int hi = lane >> 4;

  int cur = 0;
  for (int k0 = 0; k0 < K; k0 += TBK) {
    __syncthreads();
    if (k0 + TBK < K) {                       // stage next tile into alt buffer
      stageA(As[cur ^ 1], k0 + TBK);
      stageB(Bs[cur ^ 1], k0 + TBK);
    }
    // ---- fragments per ISA 16-bit layouts ----
    // A 16x32: lane m, hi=lane>>4: elems 0-7 = K[hi*8..+7], 8-15 = K[16+hi*8..+7]
    // B 32x16: lane n, hi=lane>>4: elems 0-7 = K[hi*8..+7], 8-15 = K[16+hi*8..+7]
    FragU a, bf[4];
    a.q[0] = *(const uint4*)&As[cur][ml * ROWP + hi * 8];
    a.q[1] = *(const uint4*)&As[cur][ml * ROWP + 16 + hi * 8];
#pragma unroll
    for (int j = 0; j < 4; ++j) {
      const int nl = (j << 4) + (lane & 15);
      bf[j].q[0] = *(const uint4*)&Bs[cur][nl * ROWP + hi * 8];
      bf[j].q[1] = *(const uint4*)&Bs[cur][nl * ROWP + 16 + hi * 8];
    }
#pragma unroll
    for (int j = 0; j < 4; ++j)
      acc[j] = __builtin_amdgcn_wmma_f32_16x16x32_bf16(
          false, a.v, false, bf[j].v, (short)0, acc[j], false, false);
    cur ^= 1;
  }

  // ---- store: C f32 layout: VGPR r -> M = r + (lane>>4)*8, N = lane&15 ----
  const int colL  = lane & 15;
  const int rowOf = (lane >> 4) * 8;
#pragma unroll
  for (int j = 0; j < 4; ++j) {
    int gn = n0 + (j << 4) + colL;
    if (gn >= N) continue;
    float bv = bias ? bias[gn] : 0.0f;
#pragma unroll
    for (int r = 0; r < 8; ++r) {
      int gm = m0 + (wave << 4) + rowOf + r;
      if (fullM || gm < M) C[(long long)gm * ldc + gn] = acc[j][r] + bv;
    }
  }
}

// --------------------------- conv1: K=3, pure VALU --------------------------
__global__ void k_conv1(const float* __restrict__ x, const float* __restrict__ w,
                        float* __restrict__ out, int R) {
  int idx = blockIdx.x * blockDim.x + threadIdx.x;
  if (idx >= R * 128) return;
  int r = idx >> 7, o = idx & 127;
  const float* xr = x + r * 3;
  out[idx] = w[o * 3 + 0] * xr[0] + w[o * 3 + 1] * xr[1] + w[o * 3 + 2] * xr[2];
}

// ----------------- BN (train-mode) stats: per-channel sum/sumsq -------------
__global__ __launch_bounds__(256) void k_stats(const float* __restrict__ X, int lda,
                                               int R, int C, int rowsPer,
                                               float* __restrict__ sums) {
  __shared__ float s0[1024];
  __shared__ float s1[1024];
  int tid = threadIdx.x;
  for (int c = tid; c < C; c += 256) { s0[c] = 0.f; s1[c] = 0.f; }
  __syncthreads();
  int r0 = blockIdx.x * rowsPer;
  int r1 = r0 + rowsPer; if (r1 > R) r1 = R;
  for (int r = r0; r < r1; ++r)
    for (int c = tid; c < C; c += 256) {          // channel c owned by one thread
      float v = X[(long long)r * lda + c];
      s0[c] += v; s1[c] += v * v;
    }
  __syncthreads();
  for (int c = tid; c < C; c += 256) {
    atomicAdd(&sums[c], s0[c]);
    atomicAdd(&sums[1024 + c], s1[c]);
  }
}

__global__ void k_bn_finalize(const float* __restrict__ sums,
                              const float* __restrict__ g, const float* __restrict__ b,
                              int R, int C, float* __restrict__ scale,
                              float* __restrict__ shift) {
  int c = blockIdx.x * blockDim.x + threadIdx.x;
  if (c >= C) return;
  float inv = 1.0f / (float)R;
  float m = sums[c] * inv;
  float var = sums[1024 + c] * inv - m * m;
  float sc = g[c] * rsqrtf(var + 1e-5f);
  scale[c] = sc;
  shift[c] = b[c] - m * sc;
}

// out = (res? res: 0) + act(scale*x + shift);  act: 0 none, 1 relu, 2 leaky0.2
__global__ void k_apply(const float* __restrict__ X, int ldx,
                        const float* __restrict__ scale, const float* __restrict__ shift,
                        const float* __restrict__ res, int ldr,
                        float* __restrict__ out, int ldo,
                        int R, int C, int act) {
  long long idx = (long long)blockIdx.x * blockDim.x + threadIdx.x;
  if (idx >= (long long)R * C) return;
  int r = (int)(idx / C), c = (int)(idx % C);
  float v = scale[c] * X[(long long)r * ldx + c] + shift[c];
  if (act == 1)      v = fmaxf(v, 0.0f);
  else if (act == 2) v = v > 0.0f ? v : 0.2f * v;
  if (res) v += res[(long long)r * ldr + c];
  out[(long long)r * ldo + c] = v;
}

// ------------------------- softmax over last dim, in place ------------------
__global__ __launch_bounds__(256) void k_softmax(float* __restrict__ att, int n) {
  float* row = att + (long long)blockIdx.x * n;
  __shared__ float red[256];
  int tid = threadIdx.x;
  float mx = -3.4e38f;
  for (int i = tid; i < n; i += 256) mx = fmaxf(mx, row[i]);
  red[tid] = mx; __syncthreads();
  for (int s = 128; s > 0; s >>= 1) { if (tid < s) red[tid] = fmaxf(red[tid], red[tid + s]); __syncthreads(); }
  mx = red[0]; __syncthreads();
  float sum = 0.0f;
  for (int i = tid; i < n; i += 256) { float e = __expf(row[i] - mx); row[i] = e; sum += e; }
  red[tid] = sum; __syncthreads();
  for (int s = 128; s > 0; s >>= 1) { if (tid < s) red[tid] += red[tid + s]; __syncthreads(); }
  float inv = 1.0f / red[0];
  for (int i = tid; i < n; i += 256) row[i] *= inv;
}

// column sums of att (per batch):  cs[m] = sum_n att[n][m]
__global__ void k_colsum(const float* __restrict__ att, float* __restrict__ cs, int n) {
  int m = blockIdx.x * blockDim.x + threadIdx.x;
  const float* base = att + (long long)blockIdx.y * n * n;
  float s = 0.0f;
  for (int r = 0; r < n; ++r) s += base[(long long)r * n + m];
  cs[blockIdx.y * n + m] = s;
}

// In-place:  att_new[m][n] = att_old[n][m] / (1e-9 + cs[m])   (renorm + transpose)
__global__ __launch_bounds__(1024) void k_renorm_tr(float* __restrict__ att,
                                                    const float* __restrict__ cs,
                                                    int n, int nt) {
  __shared__ float sA[32][33];
  __shared__ float sB[32][33];
  float* base = att + (long long)blockIdx.y * n * n;
  const float* c = cs + blockIdx.y * n;
  int idx = blockIdx.x, i = 0;
  while (idx >= nt - i) { idx -= nt - i; ++i; }
  int j = i + idx;
  int tx = threadIdx.x, ty = threadIdx.y;
  int r0 = i * 32, c0 = j * 32;
  sA[ty][tx] = base[(long long)(r0 + ty) * n + c0 + tx];
  if (i != j) sB[ty][tx] = base[(long long)(c0 + ty) * n + r0 + tx];
  __syncthreads();
  if (i == j) {
    base[(long long)(r0 + ty) * n + c0 + tx] = sA[tx][ty] / (1e-9f + c[r0 + ty]);
  } else {
    base[(long long)(r0 + ty) * n + c0 + tx] = sB[tx][ty] / (1e-9f + c[r0 + ty]);
    base[(long long)(c0 + ty) * n + r0 + tx] = sA[tx][ty] / (1e-9f + c[c0 + ty]);
  }
}

// pooled[b][c] = max_n leaky0.2(scale[c]*X[b,n,c] + shift[c])
__global__ void k_pool(const float* __restrict__ X, const float* __restrict__ scale,
                       const float* __restrict__ shift, float* __restrict__ out,
                       int Npts, int C) {
  int c = blockIdx.x * blockDim.x + threadIdx.x;
  int b = blockIdx.y;
  const float* base = X + (long long)b * Npts * C;
  float sc = scale[c], sh = shift[c];
  float m = -3.4e38f;
  for (int nn = 0; nn < Npts; ++nn) {
    float v = sc * base[(long long)nn * C + c] + sh;
    v = v > 0.0f ? v : 0.2f * v;
    m = fmaxf(m, v);
  }
  out[b * C + c] = m;
}

// ---------------------------------------------------------------------------
extern "C" void kernel_launch(void* const* d_in, const int* in_sizes, int n_in,
                              void* d_out, int out_size, void* d_ws, size_t ws_size,
                              hipStream_t stream) {
  (void)in_sizes; (void)n_in; (void)out_size; (void)ws_size;

  const int B = 16, NP = 2048, R = B * NP;          // 32768 rows, channel-last
  typedef long long ll;

  const float* x       = (const float*)d_in[0];
  const float* conv1_w = (const float*)d_in[1];
  const float* bn1_g   = (const float*)d_in[2];
  const float* bn1_b   = (const float*)d_in[3];
  const float* conv2_w = (const float*)d_in[4];
  const float* bn2_g   = (const float*)d_in[5];
  const float* bn2_b   = (const float*)d_in[6];
  const float* sa_qk_w = (const float*)d_in[7];     // (4,32,128)
  const float* sa_v_w  = (const float*)d_in[8];     // (4,128,128)
  const float* sa_v_b  = (const float*)d_in[9];
  const float* sa_t_w  = (const float*)d_in[10];
  const float* sa_t_b  = (const float*)d_in[11];
  const float* sa_bn_g = (const float*)d_in[12];
  const float* sa_bn_b = (const float*)d_in[13];
  const float* fuse_w  = (const float*)d_in[14];    // (1024,512)
  const float* bn5_g   = (const float*)d_in[15];
  const float* bn5_b   = (const float*)d_in[16];
  const float* lin1_w  = (const float*)d_in[17];    // (512,1024)
  const float* bn6_g   = (const float*)d_in[18];
  const float* bn6_b   = (const float*)d_in[19];
  const float* lin2_w  = (const float*)d_in[20];    // (256,512)
  const float* lin2_b  = (const float*)d_in[21];
  const float* bn7_g   = (const float*)d_in[22];
  const float* bn7_b   = (const float*)d_in[23];
  const float* lin3_w  = (const float*)d_in[24];    // (40,256)
  const float* lin3_b  = (const float*)d_in[25];
  float* out = (float*)d_out;                       // (16,40)

  // -------- workspace carve (256B aligned); total ~541 MB --------
  char* base = (char*)d_ws;
  size_t off = 0;
  auto carve = [&](size_t bytes) { char* p = base + off; off = (off + bytes + 255) & ~(size_t)255; return p; };
  float* SUMS   = (float*)carve(2048 * 4);
  float* SCALE  = (float*)carve(1024 * 4);
  float* SHIFT  = (float*)carve(1024 * 4);
  float* COLSUM = (float*)carve((size_t)R * 4);
  float* H      = (float*)carve((size_t)R * 128 * 4);
  float* V      = (float*)carve((size_t)R * 128 * 4);
  float* XR     = (float*)carve((size_t)R * 128 * 4);
  float* XT     = (float*)carve((size_t)R * 128 * 4);
  float* P      = (float*)carve((size_t)R * 32 * 4);
  float* FEATS  = (float*)carve((size_t)R * 512 * 4);
  float* ATT    = (float*)carve((size_t)B * NP * NP * 4);
  float* FUSED  = (float*)carve((size_t)R * 1024 * 4);
  float* POOL   = (float*)carve((size_t)B * 1024 * 4);
  float* TA     = (float*)carve((size_t)B * 512 * 4);
  float* TB     = (float*)carve((size_t)B * 512 * 4);
  float* TC     = (float*)carve((size_t)B * 256 * 4);
  float* TD     = (float*)carve((size_t)B * 256 * 4);

  auto gemm = [&](const float* A, int lda, ll aS, const float* Bm, int ldbK, ll bS,
                  int bKN, const float* bias, float* C, int ldc, ll cS,
                  int M, int N, int K, int batch) {
    dim3 g((N + TBN - 1) / TBN, (M + TBM - 1) / TBM, batch);
    k_gemm<<<g, 128, 0, stream>>>(A, lda, aS, Bm, ldbK, bS, bKN, bias, C, ldc, cS, M, N, K);
  };
  auto bn = [&](const float* X, int lda, int Rr, int C, const float* g, const float* b) {
    hipMemsetAsync(SUMS, 0, 2048 * sizeof(float), stream);
    int sg = Rr < 64 ? Rr : 64;
    int rowsPer = (Rr + sg - 1) / sg;
    k_stats<<<sg, 256, 0, stream>>>(X, lda, Rr, C, rowsPer, SUMS);
    k_bn_finalize<<<(C + 255) / 256, 256, 0, stream>>>(SUMS, g, b, Rr, C, SCALE, SHIFT);
  };
  auto apply = [&](const float* X, int ldx, const float* res, int ldr,
                   float* O, int ldo, int Rr, int C, int act) {
    ll tot = (ll)Rr * C;
    k_apply<<<(unsigned)((tot + 255) / 256), 256, 0, stream>>>(X, ldx, SCALE, SHIFT, res, ldr, O, ldo, Rr, C, act);
  };

  // -------- stem: conv1 -> BN -> ReLU -> conv2 -> BN -> ReLU --------
  k_conv1<<<(R * 128 + 255) / 256, 256, 0, stream>>>(x, conv1_w, XT, R);
  bn(XT, 128, R, 128, bn1_g, bn1_b);
  apply(XT, 128, nullptr, 0, XR, 128, R, 128, 1);
  gemm(XR, 128, 0, conv2_w, 0, 0, 0, nullptr, V, 128, 0, R, 128, 128, 1);
  bn(V, 128, R, 128, bn2_g, bn2_b);
  apply(V, 128, nullptr, 0, H, 128, R, 128, 1);

  // -------- 4 self-attention layers (q==k: energy = P Pᵀ per batch) --------
  for (int i = 0; i < 4; ++i) {
    const float* Hin = (i == 0) ? H : (FEATS + (i - 1) * 128);
    int ldh = (i == 0) ? 128 : 512;
    gemm(Hin, ldh, 0, sa_qk_w + (ll)i * 32 * 128, 0, 0, 0, nullptr, P, 32, 0, R, 32, 128, 1);
    gemm(Hin, ldh, 0, sa_v_w + (ll)i * 128 * 128, 0, 0, 0, sa_v_b + i * 128, V, 128, 0, R, 128, 128, 1);
    // energy[b] = P_b (2048x32) x P_bᵀ  — reuse P as the "weights" operand
    gemm(P, 32, (ll)NP * 32, P, 0, (ll)NP * 32, 0, nullptr,
         ATT, NP, (ll)NP * NP, NP, NP, 32, B);
    k_softmax<<<B * NP, 256, 0, stream>>>(ATT, NP);
    k_colsum<<<dim3(NP / 256, B), 256, 0, stream>>>(ATT, COLSUM, NP);
    k_renorm_tr<<<dim3((64 * 65) / 2, B), dim3(32, 32), 0, stream>>>(ATT, COLSUM, NP, 64);
    // x_r = attᵀ_scaled (2048x2048) x V_b (2048x128), B as [K,N] activations
    gemm(ATT, NP, (ll)NP * NP, V, 128, (ll)NP * 128, 1, nullptr,
         XR, 128, (ll)NP * 128, NP, 128, NP, B);
    gemm(XR, 128, 0, sa_t_w + (ll)i * 128 * 128, 0, 0, 0, sa_t_b + i * 128, XT, 128, 0, R, 128, 128, 1);
    bn(XT, 128, R, 128, sa_bn_g + i * 128, sa_bn_b + i * 128);
    // out = Hin + relu(bn(x_t)), written into its concat slice (ld 512)
    apply(XT, 128, Hin, ldh, FEATS + i * 128, 512, R, 128, 1);
  }

  // -------- fuse -> BN -> leaky -> global max pool --------
  gemm(FEATS, 512, 0, fuse_w, 0, 0, 0, nullptr, FUSED, 1024, 0, R, 1024, 512, 1);
  bn(FUSED, 1024, R, 1024, bn5_g, bn5_b);
  k_pool<<<dim3(1024 / 256, B), 256, 0, stream>>>(FUSED, SCALE, SHIFT, POOL, NP, 1024);

  // -------- head --------
  gemm(POOL, 1024, 0, lin1_w, 0, 0, 0, nullptr, TA, 512, 0, B, 512, 1024, 1);
  bn(TA, 512, B, 512, bn6_g, bn6_b);
  apply(TA, 512, nullptr, 0, TB, 512, B, 512, 1);
  gemm(TB, 512, 0, lin2_w, 0, 0, 0, lin2_b, TC, 256, 0, B, 256, 512, 1);
  bn(TC, 256, B, 256, bn7_g, bn7_b);
  apply(TC, 256, nullptr, 0, TD, 256, B, 256, 1);
  gemm(TD, 256, 0, lin3_w, 0, 0, 0, lin3_b, out, 40, 0, B, 40, 256, 1);
}
